// NeuralCDMModel_26654567038997
// MI455X (gfx1250) — compile-verified, matrix operand order checked by reading, not run on hardware
//
#include <hip/hip_runtime.h>
#include <hip/hip_bf16.h>
#include <math.h>

#define NUM_KP 100
#define KD 128
#define HID 256
#define HID2 128
#define KATT 20
#define SPB 8                      // students per block
#define ROWS (SPB * KATT)          // 160 rows of the attention GEMM
#define MTILES (ROWS / 16)         // 10
#define NTILES (HID / 16)          // 16

typedef __attribute__((ext_vector_type(16))) __bf16 v16bf;
typedef __attribute__((ext_vector_type(8)))  float  v8f;

__device__ __forceinline__ unsigned int pack_bf16x2(float a, float b) {
    unsigned int ua = __float_as_uint(a);
    unsigned int ub = __float_as_uint(b);
    ua += 0x7FFFu + ((ua >> 16) & 1u);   // round-to-nearest-even
    ub += 0x7FFFu + ((ub >> 16) & 1u);
    return (ua >> 16) | (ub & 0xFFFF0000u);
}

__device__ __forceinline__ float bf16_lo(unsigned int p) {
    return __uint_as_float(p << 16);
}
__device__ __forceinline__ float bf16_hi(unsigned int p) {
    return __uint_as_float(p & 0xFFFF0000u);
}

__device__ __forceinline__ float fast_tanh(float x) {
#if __has_builtin(__builtin_amdgcn_tanhf)
    return __builtin_amdgcn_tanhf(x);                 // native v_tanh_f32
#else
    const float e = __expf(2.0f * x);
    return 1.0f - 2.0f * __builtin_amdgcn_rcpf(e + 1.0f);
#endif
}

__device__ __forceinline__ float fast_rcp(float x) {
    return __builtin_amdgcn_rcpf(x);                  // v_rcp_f32
}

// ---- prep: W1 (KD,HID) fp32 -> W1T[n][kh] bf16x2, n=0..255, kh=0..63 -------
__global__ __launch_bounds__(256) void prep_w1_kernel(
    const float* __restrict__ W1, unsigned int* __restrict__ W1T)
{
    const int e  = blockIdx.x * 256 + threadIdx.x;    // 0..16383
    const int n  = e >> 6;
    const int kh = e & 63;
    W1T[e] = pack_bf16x2(W1[(2 * kh) * HID + n], W1[(2 * kh + 1) * HID + n]);
}

__global__ __launch_bounds__(256) void ncdm_fused_kernel(
    const int*   __restrict__ sidx,   // (B)
    const int*   __restrict__ kpi,    // (B,K)
    const int*   __restrict__ kpl,    // (B)
    const float* __restrict__ emb,    // (NS, NUM_KP, KD)
    const unsigned int* __restrict__ W1T, // (HID, KD/2) pre-packed bf16x2
    const float* __restrict__ b1,     // (HID)
    const float* __restrict__ w2,     // (HID)
    const float* __restrict__ b2,     // (1)
    const float* __restrict__ M1,     // (KD, HID)
    const float* __restrict__ mb1,    // (HID)
    const float* __restrict__ M2,     // (HID, HID2)
    const float* __restrict__ mb2,    // (HID2)
    const float* __restrict__ M3,     // (HID2)
    const float* __restrict__ mb3,    // (1)
    float* __restrict__ out_scores,   // (B)
    float* __restrict__ out_att)      // (B,K)
{
    __shared__ unsigned int sVec[ROWS * 64];   // masked vecs, bf16x2  (40960 B)
    __shared__ unsigned int sSlab[16 * 64];    // W1 N-slab (pre-packed) (4096 B)
    __shared__ float sLogit[ROWS];             // attention logits      (640 B)
    __shared__ float sAtt[ROWS];               // attention weights     (640 B)
    __shared__ float sW[SPB * KD];             // weighted sums         (4096 B)
    __shared__ float sH1[SPB * HID];           // hidden 1              (8192 B)
    __shared__ float sH2[SPB * HID2];          // hidden 2              (4096 B)

    const int t      = threadIdx.x;
    const int wave   = t >> 5;
    const int lane   = t & 31;
    const int hi     = lane >> 4;
    const int ln     = lane & 15;
    const int blockS = blockIdx.x * SPB;

    // ---------------- Phase 0: masked gather emb -> LDS as bf16 -------------
    {
        const int gs  = blockS + wave;
        const int st  = sidx[gs];
        const int len = kpl[gs];
        const size_t sbase = (size_t)st * (size_t)(NUM_KP * KD);
        for (int k = 0; k < KATT; ++k) {
            unsigned int p0 = 0u, p1 = 0u;
            if (k < len) {  // skip masked rows: halves average HBM gather traffic
                const int kp = kpi[gs * KATT + k];
                const float4 v = *(const float4*)(emb + sbase + (size_t)kp * KD + lane * 4);
                p0 = pack_bf16x2(v.x, v.y);
                p1 = pack_bf16x2(v.z, v.w);
            }
            const int row = wave * KATT + k;
            sVec[row * 64 + lane * 2]     = p0;
            sVec[row * 64 + lane * 2 + 1] = p1;
        }
    }
    __syncthreads();

    // -------- Phase 1: logits = tanh(vecs @ W1 + b1) @ w2 + b2 via WMMA -----
    // Wave w owns M-tiles {w} and (w<2) {w+8}. A fragments cached in VGPRs
    // across the entire N loop; per-lane partial logits accumulated in regs;
    // single cross-lane butterfly per M-tile at the end.
    union Frag { unsigned int u[8]; v16bf v; };
    Frag A1[4], A2[4];
    const int  mt1  = wave;
    const bool has2 = (wave < MTILES - 8);
    const int  mt2  = wave + 8;

    #pragma unroll
    for (int kk = 0; kk < 4; ++kk) {
        #pragma unroll
        for (int q = 0; q < 8; ++q) {
            const int koff = ((q & 4) ? 16 : 0) + ((q & 3) << 1) + (hi << 3);
            const int kh   = (kk * 32 + koff) >> 1;
            A1[kk].u[q] = sVec[(mt1 * 16 + ln) * 64 + kh];
        }
    }
    if (has2) {
        #pragma unroll
        for (int kk = 0; kk < 4; ++kk) {
            #pragma unroll
            for (int q = 0; q < 8; ++q) {
                const int koff = ((q & 4) ? 16 : 0) + ((q & 3) << 1) + (hi << 3);
                const int kh   = (kk * 32 + koff) >> 1;
                A2[kk].u[q] = sVec[(mt2 * 16 + ln) * 64 + kh];
            }
        }
    }

    float accA[8] = {}, accB[8] = {};

    for (int nt = 0; nt < NTILES; ++nt) {
        // stage pre-packed slab: one contiguous 4KB block, b128 per thread
        *(uint4*)(sSlab + t * 4) = *(const uint4*)(W1T + nt * 1024 + t * 4);
        __syncthreads();

        const float b1v = b1[nt * 16 + ln];
        const float w2v = w2[nt * 16 + ln];

        Frag Bm[4];
        #pragma unroll
        for (int kk = 0; kk < 4; ++kk) {
            #pragma unroll
            for (int q = 0; q < 8; ++q) {
                const int koff = ((q & 4) ? 16 : 0) + ((q & 3) << 1) + (hi << 3);
                Bm[kk].u[q] = sSlab[ln * 64 + ((kk * 32 + koff) >> 1)];
            }
        }

        v8f c = {};
        #pragma unroll
        for (int kk = 0; kk < 4; ++kk)
            c = __builtin_amdgcn_wmma_f32_16x16x32_bf16(
                    false, A1[kk].v, false, Bm[kk].v, (short)0, c, false, false);
        #pragma unroll
        for (int r = 0; r < 8; ++r)
            accA[r] += fast_tanh(c[r] + b1v) * w2v;

        if (has2) {   // wave-uniform: EXEC all 1s inside
            v8f c2 = {};
            #pragma unroll
            for (int kk = 0; kk < 4; ++kk)
                c2 = __builtin_amdgcn_wmma_f32_16x16x32_bf16(
                        false, A2[kk].v, false, Bm[kk].v, (short)0, c2, false, false);
            #pragma unroll
            for (int r = 0; r < 8; ++r)
                accB[r] += fast_tanh(c2[r] + b1v) * w2v;
        }
        __syncthreads();
    }

    // single butterfly reduction per M-tile (C layout: lanes 0-15 -> M=r,
    // lanes 16-31 -> M=r+8; N = lane&15)
    {
        const float b2v = b2[0];
        #pragma unroll
        for (int r = 0; r < 8; ++r) {
            float x = accA[r];
            x += __shfl_xor(x, 1);
            x += __shfl_xor(x, 2);
            x += __shfl_xor(x, 4);
            x += __shfl_xor(x, 8);
            if (ln == 0) sLogit[mt1 * 16 + hi * 8 + r] = x + b2v;
        }
        if (has2) {
            #pragma unroll
            for (int r = 0; r < 8; ++r) {
                float x = accB[r];
                x += __shfl_xor(x, 1);
                x += __shfl_xor(x, 2);
                x += __shfl_xor(x, 4);
                x += __shfl_xor(x, 8);
                if (ln == 0) sLogit[mt2 * 16 + hi * 8 + r] = x + b2v;
            }
        }
    }
    __syncthreads();

    // ---------------- Phase 2: softmax over K (then mask) -------------------
    if (t < SPB) {
        const int gs  = blockS + t;
        const int len = kpl[gs];
        float mx = -1e30f;
        #pragma unroll
        for (int k = 0; k < KATT; ++k) mx = fmaxf(mx, sLogit[t * KATT + k]);
        float sum = 0.f;
        #pragma unroll
        for (int k = 0; k < KATT; ++k) sum += __expf(sLogit[t * KATT + k] - mx);
        const float inv = fast_rcp(sum);
        #pragma unroll
        for (int k = 0; k < KATT; ++k) {
            const float a = __expf(sLogit[t * KATT + k] - mx) * inv *
                            ((k < len) ? 1.f : 0.f);
            sAtt[t * KATT + k] = a;
            out_att[(size_t)gs * KATT + k] = a;
        }
    }
    __syncthreads();

    // ---------------- Phase 3: weighted[s,d] = sum_k att * vecs -------------
    #pragma unroll
    for (int i = 0; i < 4; ++i) {
        const int o = t + 256 * i;            // 0..1023
        const int s = o >> 7;
        const int d = o & 127;
        float acc = 0.f;
        #pragma unroll
        for (int k = 0; k < KATT; ++k) {
            const unsigned int p = sVec[(s * KATT + k) * 64 + (d >> 1)];
            const float v = (d & 1) ? bf16_hi(p) : bf16_lo(p);
            acc += sAtt[s * KATT + k] * v;
        }
        sW[o] = acc;
    }
    __syncthreads();

    // ---------------- Phase 4: h1 = relu(weighted @ M1 + mb1) ---------------
    {
        const int n = t;                      // 0..255
        float acc[SPB] = {};
        for (int d = 0; d < KD; ++d) {
            const float m = M1[d * HID + n];
            #pragma unroll
            for (int s = 0; s < SPB; ++s) acc[s] += sW[s * KD + d] * m;
        }
        const float bb = mb1[n];
        #pragma unroll
        for (int s = 0; s < SPB; ++s) sH1[s * HID + n] = fmaxf(acc[s] + bb, 0.f);
    }
    __syncthreads();

    // ---------------- Phase 5: h2 = relu(h1 @ M2 + mb2) ---------------------
    {
        const int n2 = t & 127;
        const int s0 = (t >> 7) * 4;
        float acc[4] = {};
        for (int d = 0; d < HID; ++d) {
            const float m = M2[d * HID2 + n2];
            #pragma unroll
            for (int s = 0; s < 4; ++s) acc[s] += sH1[(s0 + s) * HID + d] * m;
        }
        const float bb = mb2[n2];
        #pragma unroll
        for (int s = 0; s < 4; ++s) sH2[(s0 + s) * HID2 + n2] = fmaxf(acc[s] + bb, 0.f);
    }
    __syncthreads();

    // ---------------- Phase 6: scores = sigmoid(h2 @ M3 + mb3) --------------
    if (t < SPB) {
        float acc = 0.f;
        for (int d = 0; d < HID2; ++d) acc += sH2[t * HID2 + d] * M3[d];
        const float z = acc + mb3[0];
        out_scores[blockS + t] = fast_rcp(1.f + __expf(-z));
    }
}

extern "C" void kernel_launch(void* const* d_in, const int* in_sizes, int n_in,
                              void* d_out, int out_size, void* d_ws, size_t ws_size,
                              hipStream_t stream) {
    const int*   sidx = (const int*)  d_in[0];
    const int*   kpi  = (const int*)  d_in[1];
    const int*   kpl  = (const int*)  d_in[2];
    const float* emb  = (const float*)d_in[3];
    const float* W1   = (const float*)d_in[4];
    const float* b1   = (const float*)d_in[5];
    const float* w2   = (const float*)d_in[6];
    const float* b2   = (const float*)d_in[7];
    const float* M1   = (const float*)d_in[8];
    const float* mb1  = (const float*)d_in[9];
    const float* M2   = (const float*)d_in[10];
    const float* mb2  = (const float*)d_in[11];
    const float* M3   = (const float*)d_in[12];
    const float* mb3  = (const float*)d_in[13];

    const int B = in_sizes[0];                 // 16384
    float* out_scores = (float*)d_out;         // (B)
    float* out_att    = out_scores + B;        // (B,K)

    unsigned int* W1T = (unsigned int*)d_ws;   // 64 KB pre-packed bf16 W1

    prep_w1_kernel<<<dim3(64), dim3(256), 0, stream>>>(W1, W1T);

    dim3 grid(B / SPB), block(256);
    ncdm_fused_kernel<<<grid, block, 0, stream>>>(
        sidx, kpi, kpl, emb, W1T, b1, w2, b2, M1, mb1, M2, mb2, M3, mb3,
        out_scores, out_att);
}
